// ALSHConv_11390253269180
// MI455X (gfx1250) — matrix operand order repaired and checked by reading, not compiled, not run on hardware
//
#include <hip/hip_runtime.h>
#include <stddef.h>

// ---------------------------------------------------------------------------
// ALSH conv for MI455X (gfx1250).
//  - main conv: bf16 WMMA implicit GEMM, NHWC-padded activations, async
//    global->LDS staging (ASYNCcnt) with double buffering.
//  - hash/vote/mask pipeline in f32 VALU.
// Fixed shapes: x[32,64,128,128] f32, kernels[256,64,3,3] f32,
//               a[8,67,3,3] f32, b[8] f32, table_size=512, R=4, U=0.99.
// ---------------------------------------------------------------------------

#define NIMG   32
#define CIN    64
#define HW     128
#define OCH    256
#define KK     576          // 64*9
#define NH     8
#define TSZ    512
#define PADW   130          // 128 + 1 halo each side
#define AFLAT  603          // 67*9

typedef __bf16 bf16x16 __attribute__((ext_vector_type(16)));
typedef float  f32x8   __attribute__((ext_vector_type(8)));

union Frag { bf16x16 v; uint4 q[2]; };

__device__ __forceinline__ unsigned short f2bf(float f) {
  unsigned int u = __float_as_uint(f);
  u += 0x7fffu + ((u >> 16) & 1u);       // round to nearest even
  return (unsigned short)(u >> 16);
}
__device__ __forceinline__ float bf2f(unsigned short h) {
  return __uint_as_float(((unsigned int)h) << 16);
}

// --------------------------- bulk zero (xt incl. halo) ---------------------
__global__ __launch_bounds__(256) void k_zero32(unsigned int* p, size_t n) {
  size_t i = (size_t)blockIdx.x * 256 + threadIdx.x;
  if (i < n) p[i] = 0u;
}

__global__ void k_zero(int* p, int n) {
  int i = blockIdx.x * blockDim.x + threadIdx.x;
  if (i < n) p[i] = 0;
}

// ---------------- x (NCHW f32) -> xt (padded NHWC bf16) --------------------
// block: one (image n, input row iy); LDS-tiled transpose.
__global__ __launch_bounds__(256) void k_xt(const float* __restrict__ x,
                                            unsigned short* __restrict__ xt) {
  __shared__ unsigned short tile[128 * 68];   // [ix][c], pitch 68
  const int tid = threadIdx.x;
  const int iy  = blockIdx.x;                 // 0..127 (input row)
  const int n   = blockIdx.y;
  const float* src = x + (size_t)n * CIN * HW * HW + (size_t)iy * HW;
#pragma unroll
  for (int i = 0; i < 32; ++i) {              // 64 ch x 128 px
    int e = tid + i * 256;
    int c = e >> 7, ix = e & 127;
    tile[ix * 68 + c] = f2bf(src[(size_t)c * (HW * HW) + ix]);
  }
  __syncthreads();
  // write NHWC records: pixel (iy+1, ix+1), 64 ch contiguous (128B/pixel)
  unsigned short* dst = xt + ((size_t)(n * PADW + iy + 1) * PADW + 1) * CIN;
#pragma unroll
  for (int i = 0; i < 8; ++i) {               // 2048 chunks of 4 ushorts
    int e = tid + i * 256;
    int p = e >> 4, c4 = e & 15;
    *(uint2*)(dst + (size_t)p * CIN + c4 * 4) =
        *(const uint2*)(tile + p * 68 + c4 * 4);
  }
}

// ---------------- per-kernel row norms + max (1 block) ---------------------
__global__ __launch_bounds__(256) void k_rownorms(const float* __restrict__ kern,
                                                  float* __restrict__ rnsq,
                                                  float* __restrict__ maxnsq) {
  __shared__ float red[256];
  int o = threadIdx.x;
  const float* kr = kern + (size_t)o * KK;
  float s = 0.f;
  for (int i = 0; i < KK; ++i) { float v = kr[i]; s += v * v; }
  rnsq[o] = s;
  red[o] = s;
  __syncthreads();
  for (int st = 128; st > 0; st >>= 1) {
    if (o < st) red[o] = fmaxf(red[o], red[o + st]);
    __syncthreads();
  }
  if (o == 0) maxnsq[0] = red[0];
}

// ---------------- bucket of every kernel under every hash ------------------
__global__ __launch_bounds__(256) void k_kbuckets(const float* __restrict__ a,
                                                  const float* __restrict__ b,
                                                  const float* __restrict__ kern,
                                                  const float* __restrict__ rnsq,
                                                  const float* __restrict__ maxnsq,
                                                  int* __restrict__ kb) {
  int h = blockIdx.x;
  int o = threadIdx.x;
  float s  = 0.99f / sqrtf(maxnsq[0]);       // U / max row norm
  float sn = s * sqrtf(rnsq[o]);
  const float* af = a + (size_t)h * AFLAT;
  const float* kr = kern + (size_t)o * KK;
  float dot = 0.f;
  for (int i = 0; i < KK; ++i) dot += af[i] * kr[i];
  dot *= s;
  float p = sn;
  for (int jj = 1; jj <= 27; ++jj) { p = p * p; dot += af[KK - 1 + jj] * p; }
  float v = floorf((dot + b[h]) * 0.25f);    // /R, R=4
  kb[h * OCH + o] = (int)fabsf(fmodf(v, (float)TSZ));
}

// ---------------- vote: hash-conv + LDS histogram (NHWC reads) -------------
__global__ __launch_bounds__(128) void k_vote(const unsigned short* __restrict__ xt,
                                              const float* __restrict__ a,
                                              const float* __restrict__ b,
                                              int* __restrict__ counts) {
  __shared__ float aS[NH * AFLAT];
  __shared__ int   hist[NH * TSZ];
  const int tid  = threadIdx.x;
  const int x_   = tid;            // pixel column 0..127
  const int y    = blockIdx.x;
  const int nImg = blockIdx.y;
  for (int i = tid; i < NH * AFLAT; i += 128) aS[i] = a[i];
  for (int i = tid; i < NH * TSZ;   i += 128) hist[i] = 0;
  __syncthreads();

  float acc[NH];
#pragma unroll
  for (int h = 0; h < NH; ++h) acc[h] = 0.f;

#pragma unroll
  for (int t = 0; t < 9; ++t) {
    int ky = t / 3, kx = t % 3;
    const unsigned short* base =
        xt + ((size_t)(nImg * PADW + y + ky) * PADW + (x_ + kx)) * CIN;
#pragma unroll
    for (int cq = 0; cq < 8; ++cq) {          // 8 chunks of 8 channels
      union { uint4 q; unsigned short u[8]; } u;
      u.q = *(const uint4*)(base + cq * 8);
#pragma unroll
      for (int cc = 0; cc < 8; ++cc) {
        float val = bf2f(u.u[cc]);
        int c = cq * 8 + cc;
#pragma unroll
        for (int h = 0; h < NH; ++h) acc[h] += val * aS[h * AFLAT + c * 9 + t];
      }
    }
  }
  // appended Q channels 64..66: 0.5 inside image, 0 in padding
#pragma unroll
  for (int t = 0; t < 9; ++t) {
    int ky = t / 3, kx = t % 3;
    int iy = y + ky, ix = x_ + kx;
    float val = (iy >= 1 && iy <= HW && ix >= 1 && ix <= HW) ? 0.5f : 0.0f;
#pragma unroll
    for (int h = 0; h < NH; ++h)
      acc[h] += val * (aS[h * AFLAT + 64 * 9 + t] +
                       aS[h * AFLAT + 65 * 9 + t] +
                       aS[h * AFLAT + 66 * 9 + t]);
  }
#pragma unroll
  for (int h = 0; h < NH; ++h) {
    float v = floorf((acc[h] + b[h]) * 0.25f);
    int bk = (int)fabsf(fmodf(v, (float)TSZ));
    atomicAdd(&hist[h * TSZ + bk], 1);
  }
  __syncthreads();
  for (int i = tid; i < NH * TSZ; i += 128) {
    int cnt = hist[i];
    if (cnt) atomicAdd(&counts[i], cnt);
  }
}

// ---------------- argmax per hash (first max wins, like jnp) ---------------
__global__ __launch_bounds__(256) void k_argmax(const int* __restrict__ counts,
                                                int* __restrict__ voted) {
  __shared__ int sv[256], si[256];
  int h = blockIdx.x, t = threadIdx.x;
  int v0 = counts[h * TSZ + t], v1 = counts[h * TSZ + t + 256];
  int bv, bi;
  if (v0 >= v1) { bv = v0; bi = t; } else { bv = v1; bi = t + 256; }
  sv[t] = bv; si[t] = bi;
  __syncthreads();
  for (int st = 128; st > 0; st >>= 1) {
    if (t < st) {
      int ov = sv[t + st], oi = si[t + st];
      if (ov > sv[t] || (ov == sv[t] && oi < si[t])) { sv[t] = ov; si[t] = oi; }
    }
    __syncthreads();
  }
  if (t == 0) voted[h] = si[0];
}

// ---------------- active-kernel mask ---------------------------------------
__global__ __launch_bounds__(256) void k_mask(const int* __restrict__ kb,
                                              const int* __restrict__ voted,
                                              int* __restrict__ maskv) {
  int o = threadIdx.x;
  int m = 0;
#pragma unroll
  for (int h = 0; h < NH; ++h) {
    int bo = kb[h * OCH + o];
#pragma unroll
    for (int h2 = 0; h2 < NH; ++h2) m |= (bo == voted[h2]);
  }
  maskv[o] = m;
}

// ---------------- masked weights -> bf16, K reordered as tap*64+c ----------
__global__ __launch_bounds__(256) void k_pack(const float* __restrict__ kern,
                                              const int* __restrict__ maskv,
                                              unsigned short* __restrict__ wpack) {
  int idx = blockIdx.x * 256 + threadIdx.x;      // [0, 256*576)
  int o = idx / KK, k = idx % KK;
  int tap = k >> 6, c = k & 63;                  // k = tap*64 + c
  float v = maskv[o] ? kern[(size_t)o * KK + c * 9 + tap] : 0.f;
  wpack[idx] = f2bf(v);
}

// ---------------- main conv: bf16 WMMA implicit GEMM -----------------------
// block: 256 threads (8 waves). Tile: M=128 (blockIdx.z half) x N=128 (one
// image row). 18 K-steps of 32 (tap-major). Double-buffered LDS filled with
// global_load_async_to_lds_b128 (ASYNCcnt); one barrier per K-step.
#define KP        40                 // LDS pitch (ushorts): 80B rows, 16B frags
#define BUFELEMS  (128 * KP)         // 5120 ushorts per buffer
__global__ __launch_bounds__(256) void k_conv(const unsigned short* __restrict__ xt,
                                              const unsigned short* __restrict__ wpack,
                                              float* __restrict__ out) {
  // Only __shared__ object in this kernel -> LDS offset 0; byte offsets into
  // ldsB are valid VDST addresses for the async-to-LDS instruction.
  __shared__ __align__(16) unsigned short ldsB[2 * BUFELEMS];
  const int tid   = threadIdx.x;
  const int lane  = tid & 31;
  const int wave  = tid >> 5;
  const int wy    = wave >> 1;     // 0..3 -> M sub-tiles {2wy, 2wy+1}
  const int wx    = wave & 1;      // 0..1 -> N sub-tiles {4wx..4wx+3}
  const int y     = blockIdx.x;    // output row
  const int nImg  = blockIdx.y;
  const int mHalf = blockIdx.z;    // 0/1 -> out channels [0,128)/[128,256)
  const int half  = lane >> 4;
  const int l15   = lane & 15;

  f32x8 acc[2][4];
  const f32x8 zero = {0.f, 0.f, 0.f, 0.f, 0.f, 0.f, 0.f, 0.f};
#pragma unroll
  for (int mi = 0; mi < 2; ++mi)
#pragma unroll
    for (int ni = 0; ni < 4; ++ni) acc[mi][ni] = zero;

  // stage K-step jj: copy 128 pixels x 32 channels (64B each) into buffer jj&1.
  // The LDS destination pointer is passed as an (unreferenced) asm operand so
  // the shared array escapes: with the "memory" clobber the compiler must
  // assume the asm writes ldsB, keeping the ds_load B-fragments real.
  auto stage = [&](int jj) {
    const int tap = jj >> 1;
    const int ky = tap / 3, kx = tap % 3;
    const int c0 = (jj & 1) * 32;
    const unsigned bufbase = (unsigned)((jj & 1) * BUFELEMS * 2);
    const unsigned short* row =
        xt + ((size_t)(nImg * PADW + y + ky) * PADW) * CIN;
#pragma unroll
    for (int it = 0; it < 2; ++it) {
      int s = tid + it * 256;                // 0..511
      int p = s >> 2, part = s & 3;          // pixel, 16B quarter
      const unsigned short* g = row + (size_t)(p + kx) * CIN + c0 + part * 8;
      unsigned short* ldst = ldsB + (jj & 1) * BUFELEMS + p * KP + part * 8;
      unsigned loff = bufbase + (unsigned)(p * KP + part * 8) * 2;
      asm volatile("global_load_async_to_lds_b128 %0, %1, off"
                   :: "v"(loff), "v"(g), "v"(ldst) : "memory");
    }
  };

  stage(0);
  for (int j = 0; j < 18; ++j) {
    asm volatile("s_wait_asynccnt 0" ::: "memory");
    __syncthreads();                         // buffer j&1 ready for all waves
    if (j + 1 < 18) stage(j + 1);            // prefetch into the other buffer

    Frag A[2];
#pragma unroll
    for (int mi = 0; mi < 2; ++mi) {
      int mrow = mHalf * 128 + (wy * 2 + mi) * 16 + l15;
      const unsigned short* wp = wpack + (size_t)mrow * KK + j * 32 + half * 8;
      A[mi].q[0] = *reinterpret_cast<const uint4*>(wp);        // K 0-7 / 8-15
      A[mi].q[1] = *reinterpret_cast<const uint4*>(wp + 16);   // K 16-23 / 24-31
    }
    const unsigned short* buf = ldsB + (j & 1) * BUFELEMS;
    Frag B[4];
#pragma unroll
    for (int ni = 0; ni < 4; ++ni) {
      int col = (wx * 4 + ni) * 16 + l15;
      const unsigned short* bp = buf + col * KP + half * 8;    // K 0-7,16-23 / 8-15,24-31
      B[ni].q[0] = *reinterpret_cast<const uint4*>(bp);
      B[ni].q[1] = *reinterpret_cast<const uint4*>(bp + 16);
    }
#pragma unroll
    for (int mi = 0; mi < 2; ++mi)
#pragma unroll
      for (int ni = 0; ni < 4; ++ni)
        acc[mi][ni] = __builtin_amdgcn_wmma_f32_16x16x32_bf16(
            false, A[mi].v, false, B[ni].v, (short)0, acc[mi][ni], false, false);
  }

  // epilogue: C layout -> out[n][M][y][N]
#pragma unroll
  for (int mi = 0; mi < 2; ++mi) {
#pragma unroll
    for (int ni = 0; ni < 4; ++ni) {
      int N = (wx * 4 + ni) * 16 + l15;
#pragma unroll
      for (int v = 0; v < 8; ++v) {
        int M = mHalf * 128 + (wy * 2 + mi) * 16 + v + 8 * half;
        size_t idx = (((size_t)nImg * OCH + M) * HW + y) * HW + N;
        out[idx] = acc[mi][ni][v];
      }
    }
  }
}

// ---------------------------------------------------------------------------
extern "C" void kernel_launch(void* const* d_in, const int* in_sizes, int n_in,
                              void* d_out, int out_size, void* d_ws, size_t ws_size,
                              hipStream_t stream) {
  (void)in_sizes; (void)n_in; (void)out_size; (void)ws_size;
  const float* x    = (const float*)d_in[0];
  const float* kern = (const float*)d_in[1];
  const float* a    = (const float*)d_in[2];
  const float* b    = (const float*)d_in[3];
  // d_in[4] = table_size (always 512; hardcoded as TSZ)
  float* out = (float*)d_out;
  char*  ws  = (char*)d_ws;

  constexpr size_t XT_BYTES   = (size_t)NIMG * PADW * PADW * CIN * 2;  // 69.2 MB
  constexpr size_t OFF_WPACK  = XT_BYTES;                  // 256-aligned
  constexpr size_t OFF_COUNTS = OFF_WPACK + (size_t)OCH * KK * 2;
  constexpr size_t OFF_RNSQ   = OFF_COUNTS + NH * TSZ * 4;
  constexpr size_t OFF_MAXN   = OFF_RNSQ + OCH * 4;
  constexpr size_t OFF_KB     = OFF_MAXN + 256;
  constexpr size_t OFF_VOTED  = OFF_KB + NH * OCH * 4;
  constexpr size_t OFF_MASK   = OFF_VOTED + 256;

  unsigned short* xt     = (unsigned short*)(ws);
  unsigned short* wpack  = (unsigned short*)(ws + OFF_WPACK);
  int*            counts = (int*)(ws + OFF_COUNTS);
  float*          rnsq   = (float*)(ws + OFF_RNSQ);
  float*          maxnsq = (float*)(ws + OFF_MAXN);
  int*            kb     = (int*)(ws + OFF_KB);
  int*            voted  = (int*)(ws + OFF_VOTED);
  int*            maskv  = (int*)(ws + OFF_MASK);

  const size_t xtWords = XT_BYTES / 4;
  k_zero32<<<(unsigned)((xtWords + 255) / 256), 256, 0, stream>>>(
      (unsigned int*)xt, xtWords);
  k_xt<<<dim3(HW, NIMG), 256, 0, stream>>>(x, xt);
  k_zero<<<(NH * TSZ + 255) / 256, 256, 0, stream>>>(counts, NH * TSZ);
  k_rownorms<<<1, 256, 0, stream>>>(kern, rnsq, maxnsq);
  k_kbuckets<<<NH, 256, 0, stream>>>(a, b, kern, rnsq, maxnsq, kb);
  k_vote<<<dim3(HW, NIMG), 128, 0, stream>>>(xt, a, b, counts);
  k_argmax<<<NH, 256, 0, stream>>>(counts, voted);
  k_mask<<<1, 256, 0, stream>>>(kb, voted, maskv);
  k_pack<<<(OCH * KK) / 256, 256, 0, stream>>>(kern, maskv, wpack);
  k_conv<<<dim3(HW, NIMG, 2), 256, 0, stream>>>(xt, wpack, out);
}
